// PointNetSetAbstractionMsg_34634616275289
// MI455X (gfx1250) — matrix operand order repaired and checked by reading, not compiled, not run on hardware
//
#include <hip/hip_runtime.h>
#include <hip/hip_bf16.h>

typedef _Float16 h16;
typedef __attribute__((ext_vector_type(16))) _Float16 v16h;
typedef __attribute__((ext_vector_type(8)))  float    v8f;
typedef int v4i_sgpr __attribute__((ext_vector_type(4)));
typedef int v8i_sgpr __attribute__((ext_vector_type(8)));

// Problem constants (from reference)
#define BATCH 4
#define NPTS  16384
#define S     2048
#define BS    (BATCH * S)      // 8192 centroids
#define IN_CH 32
#define OUT_CH_TOTAL 320
#define BN_EPS 1e-5f

// Workspace layout (bytes). Assumes generous ws (~600MB needed).
#define OFF_FPS     ((size_t)0)                       // BS ints            = 32 KB
#define OFF_NEWXYZ  ((size_t)32768)                   // BS*3 floats        = 96 KB
#define OFF_BQ0     ((size_t)131072)                  // BS*16 ints
#define OFF_BQ1     (OFF_BQ0 + (size_t)BS*16*4)
#define OFF_BQ2     (OFF_BQ1 + (size_t)BS*32*4)
#define OFF_W16     (OFF_BQ2 + (size_t)BS*128*4)      // f16 weights, < 128 KB
#define OFF_STATS   (OFF_W16 + (size_t)131072)        // 4*128 floats
#define OFF_X       ((size_t)8  * 1024 * 1024)        // f16 activations in
#define OFF_Y       (OFF_X + (size_t)268435456)       // f16 activations out

// ---------------------------------------------------------------------------
// Tensor Data Mover: 1-D contiguous global->LDS copy (NBYTES multiple of 8).
// Builds a 2-group D# (VADDR2/3 = NULL) per CDNA5 ISA:
//   group0: [1:0]=count=1, [63:32]=lds_addr, [120:64]=global_addr, [127:126]=2
//   group1: data_size=3 (8B), tensor_dim0=tile_dim0=NBYTES/8, rest 0
// Issued by one wave; completion via TENSORcnt.
// ---------------------------------------------------------------------------
template <unsigned NBYTES>
__device__ inline void tdm_copy_1d(const void* gsrc, void* lds_dst) {
  constexpr unsigned E = NBYTES / 8;  // 8-byte elements, E < 65536
  unsigned lds = (unsigned)(size_t)lds_dst;           // low 32 bits = LDS offset
  unsigned long long ga = (unsigned long long)(size_t)gsrc;
  v4i_sgpr g0;
  g0[0] = 1;                                          // count=1, user descriptor
  g0[1] = (int)lds;                                   // lds_addr
  g0[2] = (int)(unsigned)(ga & 0xffffffffu);          // global_addr[31:0]
  g0[3] = (int)((unsigned)((ga >> 32) & 0x1ffffffu) | (2u << 30)); // [56:32]|type=2
  v8i_sgpr g1;
  g1[0] = (int)(3u << 16);                            // data_size = 8 bytes
  g1[1] = (int)((E & 0xffffu) << 16);                 // tensor_dim0[15:0]
  g1[2] = (int)(E >> 16);                             // tensor_dim0[31:16]
  g1[3] = (int)(E << 16);                             // tile_dim0 = E
  g1[4] = 0; g1[5] = 0; g1[6] = 0; g1[7] = 0;
  asm volatile("tensor_load_to_lds %0, %1" :: "s"(g0), "s"(g1) : "memory");
}

// ---------------------------------------------------------------------------
// FPS: one 1024-thread block per batch. Each thread owns 16 points in regs.
// ---------------------------------------------------------------------------
__global__ __launch_bounds__(1024) void fps_kernel(const float* __restrict__ xyz,
                                                   int* __restrict__ fps_idx) {
  const int b   = blockIdx.x;
  const int tid = threadIdx.x;
  const float* base = xyz + (size_t)b * NPTS * 3;
  float px[16], py[16], pz[16], md[16];
#pragma unroll
  for (int j = 0; j < 16; ++j) {
    int p = j * 1024 + tid;
    px[j] = base[p * 3 + 0];
    py[j] = base[p * 3 + 1];
    pz[j] = base[p * 3 + 2];
    md[j] = 1e10f;
  }
  __shared__ float sv[32];
  __shared__ int   si[32];
  __shared__ float cur[3];
  __shared__ int   curIdx;
  if (tid == 0) curIdx = 0;
  __syncthreads();
  int* out = fps_idx + b * S;
  const int lane = tid & 31, wv = tid >> 5;
  for (int s = 0; s < S; ++s) {
    if (tid == 0) {
      int ci = curIdx;
      out[s] = ci;                      // scan outputs previous 'last' (starts at 0)
      cur[0] = base[ci * 3 + 0];
      cur[1] = base[ci * 3 + 1];
      cur[2] = base[ci * 3 + 2];
    }
    __syncthreads();
    const float lx = cur[0], ly = cur[1], lz = cur[2];
    float bv = -1.f; int bi = 0;
#pragma unroll
    for (int j = 0; j < 16; ++j) {
      float dx = px[j] - lx, dy = py[j] - ly, dz = pz[j] - lz;
      float d = dx * dx + dy * dy + dz * dz;
      md[j] = fminf(md[j], d);
      int p = j * 1024 + tid;
      if (md[j] > bv) { bv = md[j]; bi = p; }   // j ascending => first-max kept
    }
    for (int off = 16; off > 0; off >>= 1) {
      float ov = __shfl_xor(bv, off, 32);
      int   oi = __shfl_xor(bi, off, 32);
      if (ov > bv || (ov == bv && oi < bi)) { bv = ov; bi = oi; }
    }
    if (lane == 0) { sv[wv] = bv; si[wv] = bi; }
    __syncthreads();
    if (tid == 0) {
      float b0 = sv[0]; int i0 = si[0];
      for (int w = 1; w < 32; ++w)
        if (sv[w] > b0 || (sv[w] == b0 && si[w] < i0)) { b0 = sv[w]; i0 = si[w]; }
      curIdx = i0;
    }
    __syncthreads();
  }
}

// ---------------------------------------------------------------------------
// Gather new_xyz into ws and into d_out (first B*S*3 floats of output).
// ---------------------------------------------------------------------------
__global__ void gather_newxyz(const float* __restrict__ xyz, const int* __restrict__ fps_idx,
                              float* __restrict__ nxyz, float* __restrict__ out) {
  int i = blockIdx.x * blockDim.x + threadIdx.x;
  if (i >= BS) return;
  int b = i / S;
  int pid = fps_idx[i];
  const float* p = xyz + ((size_t)b * NPTS + pid) * 3;
  float x = p[0], y = p[1], z = p[2];
  nxyz[i * 3 + 0] = x; nxyz[i * 3 + 1] = y; nxyz[i * 3 + 2] = z;
  out[i * 3 + 0]  = x; out[i * 3 + 1]  = y; out[i * 3 + 2]  = z;
}

// ---------------------------------------------------------------------------
// Ball query, all 3 radii in one pass. One wave per centroid, ballot compact.
// Ascending index order matches jnp.sort; centroid itself guarantees >=1 hit.
// ---------------------------------------------------------------------------
__global__ __launch_bounds__(256) void ball_query(const float* __restrict__ xyz,
                                                  const float* __restrict__ nxyz,
                                                  int* __restrict__ idx0,
                                                  int* __restrict__ idx1,
                                                  int* __restrict__ idx2) {
  __shared__ int buf[8][176];
  const int lane = threadIdx.x & 31;
  const int wv   = threadIdx.x >> 5;
  const int bs   = blockIdx.x * 8 + wv;
  const int b    = bs / S;
  const float cx = nxyz[bs * 3 + 0], cy = nxyz[bs * 3 + 1], cz = nxyz[bs * 3 + 2];
  const float* px = xyz + (size_t)b * NPTS * 3;
  int* b0 = buf[wv]; int* b1 = b0 + 16; int* b2 = b1 + 32;
  int c0 = 0, c1 = 0, c2 = 0;
  const float r0 = 0.01f, r1 = 0.04f, r2 = 0.16f;
  const unsigned lt = (1u << lane) - 1u;
  for (int base = 0; base < NPTS; base += 32) {
    if (c0 >= 16 && c1 >= 32 && c2 >= 128) break;
    int p = base + lane;
    float dx = px[p * 3 + 0] - cx, dy = px[p * 3 + 1] - cy, dz = px[p * 3 + 2] - cz;
    float d2 = dx * dx + dy * dy + dz * dz;
    if (c0 < 16) {
      unsigned m = (unsigned)__ballot(d2 <= r0);
      int pos = c0 + __popc(m & lt);
      if (d2 <= r0 && pos < 16) b0[pos] = p;
      c0 = min(c0 + __popc(m), 16);
    }
    if (c1 < 32) {
      unsigned m = (unsigned)__ballot(d2 <= r1);
      int pos = c1 + __popc(m & lt);
      if (d2 <= r1 && pos < 32) b1[pos] = p;
      c1 = min(c1 + __popc(m), 32);
    }
    if (c2 < 128) {
      unsigned m = (unsigned)__ballot(d2 <= r2);
      int pos = c2 + __popc(m & lt);
      if (d2 <= r2 && pos < 128) b2[pos] = p;
      c2 = min(c2 + __popc(m), 128);
    }
  }
  int f0 = b0[0], f1 = b1[0], f2 = b2[0];
  for (int i = c0 + lane; i < 16; i += 32)  b0[i] = f0;
  for (int i = c1 + lane; i < 32; i += 32)  b1[i] = f1;
  for (int i = c2 + lane; i < 128; i += 32) b2[i] = f2;
  for (int i = lane; i < 16; i += 32)  idx0[bs * 16 + i]  = b0[i];
  for (int i = lane; i < 32; i += 32)  idx1[bs * 32 + i]  = b1[i];
  for (int i = lane; i < 128; i += 32) idx2[bs * 128 + i] = b2[i];
}

// ---------------------------------------------------------------------------
// Build layer-0 input: [rel_xyz(3) | feat(32) | zero-pad to 64] as f16.
// ---------------------------------------------------------------------------
__global__ void build_layer0(const float* __restrict__ xyz, const float* __restrict__ feat,
                             const float* __restrict__ nxyz, const int* __restrict__ idxs,
                             h16* __restrict__ X0, int K) {
  int i = blockIdx.x * blockDim.x + threadIdx.x;
  if (i >= BS * K) return;
  int bs = i / K;
  int b  = bs / S;
  int pid = idxs[i];
  const float* p  = xyz + ((size_t)b * NPTS + pid) * 3;
  const float* cp = nxyz + (size_t)bs * 3;
  h16* xo = X0 + (size_t)i * 64;
  xo[0] = (h16)(p[0] - cp[0]);
  xo[1] = (h16)(p[1] - cp[1]);
  xo[2] = (h16)(p[2] - cp[2]);
  const float* f = feat + ((size_t)b * NPTS + pid) * IN_CH;
#pragma unroll
  for (int c = 0; c < IN_CH; ++c) xo[3 + c] = (h16)f[c];
#pragma unroll
  for (int c = 35; c < 64; ++c) xo[c] = (h16)0.f;
}

// ---------------------------------------------------------------------------
// Weight prep: f32 (Cout x Cin) -> f16 (Cout x CinP), zero-padded K dim.
// ---------------------------------------------------------------------------
__global__ void prep_weights(const float* __restrict__ w, h16* __restrict__ w16,
                             int cin, int cinP, int cout) {
  int i = blockIdx.x * blockDim.x + threadIdx.x;
  if (i >= cout * cinP) return;
  int c = i % cinP, o = i / cinP;
  w16[i] = (c < cin) ? (h16)w[o * cin + c] : (h16)0.f;
}

// ---------------------------------------------------------------------------
// WMMA GEMM: Y[M x COUT] = X[M x CINP] * W^T + bias (bias folded into the C
// matrix), Y stored f16 via an LDS transpose for coalesced b128 stores, and
// per-channel sum / sum^2 accumulated (for batch-global BN).
// 256 threads = 8 waves; wave = 16-row tile; block = 128 rows.
// X/W tiles are staged to LDS with the Tensor Data Mover (TENSORcnt).
// M is guaranteed to be a multiple of 128 by the host (no row guards).
// ---------------------------------------------------------------------------
template <int CINP, int COUT>
__global__ __launch_bounds__(256) void gemm_bn_kernel(
    const h16* __restrict__ X, const h16* __restrict__ W,
    const float* __restrict__ bias, h16* __restrict__ Y,
    float* __restrict__ gsum, float* __restrict__ gsqs, int M) {
  constexpr int NT = COUT / 16;
  constexpr int KT = CINP / 32;
  constexpr int MAXD = (CINP > COUT) ? CINP : COUT;
  __shared__ __align__(32) h16 sXO[128 * MAXD];   // X tile, then reused for D tile
  __shared__ __align__(32) h16 sW[COUT * CINP];
  __shared__ float sSum[COUT];
  __shared__ float sSqs[COUT];
  const int tid = threadIdx.x;
  const int lane = tid & 31;
  const int wv   = tid >> 5;
  const int blockRow = blockIdx.x * 128;
  const int r16 = lane & 15;
  const int hi  = lane >> 4;

  for (int c = tid; c < COUT; c += 256) { sSum[c] = 0.f; sSqs[c] = 0.f; }

  // Stage W (Cout x CinP) and the 128-row X tile via TDM, wave 0 only.
  if (wv == 0) {
    tdm_copy_1d<(unsigned)COUT * CINP * 2>((const void*)W, (void*)sW);
    tdm_copy_1d<(unsigned)128 * CINP * 2>((const void*)(X + (size_t)blockRow * CINP),
                                          (void*)sXO);
    __builtin_amdgcn_s_wait_tensorcnt(0);
  }

  // Bias folded into accumulator: C[m][n] = bias[n] for every row m.
  v8f acc[NT];
#pragma unroll
  for (int nt = 0; nt < NT; ++nt) {
    float bv = bias[nt * 16 + r16];
    acc[nt] = (v8f){bv, bv, bv, bv, bv, bv, bv, bv};
  }
  __syncthreads();

#pragma unroll
  for (int kt = 0; kt < KT; ++kt) {
    // A fragment: lane (hi=0) rows hold K {0..7,16..23}; (hi=1) K {8..15,24..31}
    union { v16h v; uint4 q[2]; } au;
    const h16* xrow = &sXO[(wv * 16 + r16) * CINP + kt * 32 + hi * 8];
    au.q[0] = *(const uint4*)xrow;
    au.q[1] = *(const uint4*)(xrow + 16);
#pragma unroll
    for (int nt = 0; nt < NT; ++nt) {
      // B fragment: lane = column n, (hi=0) K {0..7,16..23}; (hi=1) K {8..15,24..31}
      const h16* wrow = &sW[(nt * 16 + r16) * CINP + kt * 32 + hi * 8];
      union { v16h v; uint4 q[2]; } bu;
      bu.q[0] = *(const uint4*)wrow;
      bu.q[1] = *(const uint4*)(wrow + 16);
      v16h bf = bu.v;
      acc[nt] = __builtin_amdgcn_wmma_f32_16x16x32_f16(
          false, au.v, false, bf, (short)0, acc[nt], false, false);
    }
  }

  // Epilogue: stats + f16 D tile into LDS (reuse sXO), then coalesced stores.
  __syncthreads();  // all waves done reading sXO/sW
#pragma unroll
  for (int nt = 0; nt < NT; ++nt) {
    const int col = nt * 16 + r16;
    float s = 0.f, s2 = 0.f;
#pragma unroll
    for (int v = 0; v < 8; ++v) {
      float val = acc[nt][v];
      s += val; s2 += val * val;
      sXO[(wv * 16 + hi * 8 + v) * COUT + col] = (h16)val;
    }
    s  += __shfl_xor(s, 16, 32);
    s2 += __shfl_xor(s2, 16, 32);
    if (hi == 0) {
      atomicAdd(&sSum[col], s);
      atomicAdd(&sSqs[col], s2);
    }
  }
  __syncthreads();
  {  // coalesced b128 stores of the 128 x COUT f16 tile
    const uint4* src = (const uint4*)sXO;
    uint4* dst = (uint4*)(Y + (size_t)blockRow * COUT);
    for (int i = tid; i < 128 * COUT / 8; i += 256) dst[i] = src[i];
  }
  for (int c = tid; c < COUT; c += 256) {
    atomicAdd(&gsum[c], sSum[c]);
    atomicAdd(&gsqs[c], sSqs[c]);
  }
  (void)M;
}

// ---------------------------------------------------------------------------
// Fold BN stats into per-channel scale/shift.
// ---------------------------------------------------------------------------
__global__ void finalize_stats(const float* __restrict__ gsum, const float* __restrict__ gsqs,
                               const float* __restrict__ gamma, const float* __restrict__ beta,
                               float* __restrict__ scl, float* __restrict__ shf,
                               int cout, float invM) {
  int c = threadIdx.x;
  if (c >= cout) return;
  float mean = gsum[c] * invM;
  float var  = gsqs[c] * invM - mean * mean;
  float a = gamma[c] * rsqrtf(var + BN_EPS);
  scl[c] = a;
  shf[c] = beta[c] - mean * a;
}

// ---------------------------------------------------------------------------
// BN + ReLU + cast f16 into next layer's (padded) input buffer.
// ---------------------------------------------------------------------------
__global__ void normalize_relu(const h16* __restrict__ Y, const float* __restrict__ scl,
                               const float* __restrict__ shf, h16* __restrict__ Xn,
                               int M, int cout, int cinpN) {
  long long i = (long long)blockIdx.x * blockDim.x + threadIdx.x;
  long long total = (long long)M * cinpN;
  if (i >= total) return;
  int c = (int)(i % cinpN);
  long long row = i / cinpN;
  float v = 0.f;
  if (c < cout) {
    float y = (float)Y[row * cout + c];
    v = fmaxf(y * scl[c] + shf[c], 0.f);
  }
  Xn[i] = (h16)v;
}

// ---------------------------------------------------------------------------
// Final layer: BN + ReLU + max over K, write into output feature slab.
// ---------------------------------------------------------------------------
__global__ void pool_max(const h16* __restrict__ Y, const float* __restrict__ scl,
                         const float* __restrict__ shf, float* __restrict__ out,
                         int K, int cout, int chanOff) {
  int i = blockIdx.x * blockDim.x + threadIdx.x;
  if (i >= BS * cout) return;
  int c = i % cout, bs = i / cout;
  const float a = scl[c], sh = shf[c];
  float m = 0.f;  // relu >= 0, so 0 is a valid identity
  const h16* yp = Y + (size_t)bs * K * cout + c;
  for (int k = 0; k < K; ++k)
    m = fmaxf(m, fmaxf((float)yp[(size_t)k * cout] * a + sh, 0.f));
  out[(size_t)BS * 3 + (size_t)bs * OUT_CH_TOTAL + chanOff + c] = m;
}

// ---------------------------------------------------------------------------
// Host side
// ---------------------------------------------------------------------------
struct LC { int cin, cinP, cout; };
static const LC LCFG[3][3] = {
    {{35, 64, 32}, {32, 32, 32}, {32, 32, 64}},
    {{35, 64, 64}, {64, 64, 64}, {64, 64, 128}},
    {{35, 64, 64}, {64, 64, 96}, {96, 96, 128}},
};
static const int KS[3] = {16, 32, 128};
static const int CHAN_OFF[3] = {0, 64, 192};

static void launch_gemm(int cinP, int cout, const h16* X, const h16* W, const float* bias,
                        h16* Y, float* gsum, float* gsqs, int M, hipStream_t stream) {
  dim3 grid((M + 127) / 128);
#define GEMM_CASE(CP, CO)                                                        \
  if (cinP == CP && cout == CO) {                                                \
    gemm_bn_kernel<CP, CO><<<grid, 256, 0, stream>>>(X, W, bias, Y, gsum, gsqs, M); \
    return;                                                                      \
  }
  GEMM_CASE(64, 32)
  GEMM_CASE(32, 32)
  GEMM_CASE(32, 64)
  GEMM_CASE(64, 64)
  GEMM_CASE(64, 96)
  GEMM_CASE(64, 128)
  GEMM_CASE(96, 128)
#undef GEMM_CASE
}

extern "C" void kernel_launch(void* const* d_in, const int* in_sizes, int n_in,
                              void* d_out, int out_size, void* d_ws, size_t ws_size,
                              hipStream_t stream) {
  (void)in_sizes; (void)n_in; (void)out_size; (void)ws_size;
  const float* xyz  = (const float*)d_in[0];
  const float* feat = (const float*)d_in[1];
  char* ws = (char*)d_ws;
  int*   fps  = (int*)(ws + OFF_FPS);
  float* nxyz = (float*)(ws + OFF_NEWXYZ);
  int*   bq0  = (int*)(ws + OFF_BQ0);
  int*   bq1  = (int*)(ws + OFF_BQ1);
  int*   bq2  = (int*)(ws + OFF_BQ2);
  h16*   w16b = (h16*)(ws + OFF_W16);
  float* stats = (float*)(ws + OFF_STATS);  // [sum128 | sqs128 | scl128 | shf128]
  h16*   Xbuf = (h16*)(ws + OFF_X);
  h16*   Ybuf = (h16*)(ws + OFF_Y);
  float* out = (float*)d_out;

  fps_kernel<<<BATCH, 1024, 0, stream>>>(xyz, fps);
  gather_newxyz<<<(BS + 255) / 256, 256, 0, stream>>>(xyz, fps, nxyz, out);
  ball_query<<<BS / 8, 256, 0, stream>>>(xyz, nxyz, bq0, bq1, bq2);

  size_t woff = 0;
  for (int sc = 0; sc < 3; ++sc) {
    const int K = KS[sc];
    const int M = BS * K;   // 131072 / 262144 / 1048576 -- multiples of 128
    const int* bq = (sc == 0) ? bq0 : (sc == 1) ? bq1 : bq2;
    build_layer0<<<(M + 255) / 256, 256, 0, stream>>>(xyz, feat, nxyz, bq, Xbuf, K);
    for (int l = 0; l < 3; ++l) {
      const LC cfg = LCFG[sc][l];
      const int pi = 2 + (sc * 3 + l) * 4;
      const float* w     = (const float*)d_in[pi + 0];
      const float* bias  = (const float*)d_in[pi + 1];
      const float* gamma = (const float*)d_in[pi + 2];
      const float* beta  = (const float*)d_in[pi + 3];
      h16* w16 = w16b + woff;
      woff += (size_t)cfg.cout * cfg.cinP;
      int wcnt = cfg.cout * cfg.cinP;
      prep_weights<<<(wcnt + 255) / 256, 256, 0, stream>>>(w, w16, cfg.cin, cfg.cinP, cfg.cout);
      hipMemsetAsync(stats, 0, 2 * 128 * sizeof(float), stream);
      launch_gemm(cfg.cinP, cfg.cout, Xbuf, w16, bias, Ybuf, stats, stats + 128, M, stream);
      finalize_stats<<<1, 128, 0, stream>>>(stats, stats + 128, gamma, beta,
                                            stats + 256, stats + 384, cfg.cout, 1.f / (float)M);
      if (l < 2) {
        const int cinpN = LCFG[sc][l + 1].cinP;
        long long total = (long long)M * cinpN;
        normalize_relu<<<(unsigned)((total + 255) / 256), 256, 0, stream>>>(
            Ybuf, stats + 256, stats + 384, Xbuf, M, cfg.cout, cinpN);
      } else {
        pool_max<<<(BS * cfg.cout + 255) / 256, 256, 0, stream>>>(
            Ybuf, stats + 256, stats + 384, out, K, cfg.cout, CHAN_OFF[sc]);
      }
    }
  }
}
